// TrackAgileModuleVer2Dicision_11708080849079
// MI455X (gfx1250) — matrix operand order repaired and checked by reading, not compiled, not run on hardware
//
#include <hip/hip_runtime.h>
#include <hip/hip_bf16.h>

// ---------------------------------------------------------------------------
// Problem constants (from reference): T=128, B=2048, I=18, H=256, G=3H=768
// ---------------------------------------------------------------------------
#define GRU_T 128
#define GRU_B 2048
#define GRU_I 18
#define GRU_H 256
#define GRU_G 768

// LDS row strides (elements) chosen for conflict-free ds_load_b128:
// 264*2B = 528B = 132 DW  => lane n starts at bank 4n (mod 64): full spread.
// 40*2B  =  80B =  20 DW  => lane n starts at bank 20n (mod 64): full spread.
#define LDSW_K256 264
#define LDSW_K32  40

typedef __attribute__((ext_vector_type(16))) __bf16        v16bf;
typedef __attribute__((ext_vector_type(8)))  float         v8f;
typedef __attribute__((ext_vector_type(4)))  unsigned int  u32x4;

union Frag16 {
    u32x4          u[2];
    v16bf          v;
    unsigned short s[16];
};

__device__ __forceinline__ unsigned short f2bf(float x) {
    union { float f; unsigned int u; } c;
    c.f = x;
    unsigned int u = c.u;
    u += 0x7fffu + ((u >> 16) & 1u);   // round-to-nearest-even
    return (unsigned short)(u >> 16);
}

__device__ __forceinline__ float sig_fast(float x) {
    return 1.0f / (1.0f + __expf(-x));
}
__device__ __forceinline__ float tanh_fast(float x) {
    float e = __expf(-2.0f * x);
    return (1.0f - e) / (1.0f + e);
}

__device__ __forceinline__ v8f wmma_bf16(v16bf a, v16bf b, v8f c) {
    return __builtin_amdgcn_wmma_f32_16x16x32_bf16(
        /*neg_a=*/false, a, /*neg_b=*/false, b,
        /*c_mod=*/(short)0, c, /*reuse_a=*/false, /*reuse_b=*/false);
}

// A fragment (16x32 bf16, row-major, leading dim lda).  ISA 7.12.2:
// lane<16: row M=lane, K={k..k+7, k+16..k+23}; lane>=16: K={k+8..15, k+24..31}.
__device__ __forceinline__ v16bf load_a_frag(const unsigned short* __restrict__ base,
                                             int lda, int lane) {
    const int r  = lane & 15;
    const int hi = lane >> 4;
    const unsigned short* p = base + (size_t)r * lda + hi * 8;
    Frag16 f;
    f.u[0] = *(const u32x4*)(p);
    f.u[1] = *(const u32x4*)(p + 16);
    return f.v;
}

// B fragment (32x16 bf16) from an LDS-staged weight slice.  B[k][n] = W[n][k]
// (W rows are B columns), rows re-indexed to local slice rows, padded stride.
__device__ __forceinline__ v16bf load_b_frag_lds(const unsigned short* W,
                                                 int strideE, int rowBase,
                                                 int k0, int lane) {
    const int n  = lane & 15;
    const int hi = lane >> 4;
    const unsigned short* p = W + (rowBase + n) * strideE + k0 + hi * 8;
    Frag16 f;
    f.u[0] = *(const u32x4*)(p);
    f.u[1] = *(const u32x4*)(p + 16);
    return f.v;
}

// Layer-0 A fragment built from fp32 x_t [B,18], zero-padded to K=32.
__device__ __forceinline__ v16bf load_a_frag_x18(const float* __restrict__ X,
                                                 int m0, int lane) {
    const int r  = lane & 15;
    const int kb = (lane >> 4) * 8;
    const float* row = X + (size_t)(m0 + r) * GRU_I;
    Frag16 f;
#pragma unroll
    for (int j = 0; j < 8; ++j) {
        int k = kb + j;
        f.s[j] = (k < GRU_I) ? f2bf(row[k]) : (unsigned short)0;
    }
#pragma unroll
    for (int j = 0; j < 8; ++j) {
        int k = kb + 16 + j;
        f.s[8 + j] = (k < GRU_I) ? f2bf(row[k]) : (unsigned short)0;
    }
    return f.v;
}

// ---------------------------------------------------------------------------
// One GRU layer-step:  gates = x@Wih^T + b_ih + h@Whh^T + b_hh  ->  h_new.
// Grid (16, 8), block 256 threads = 8 waves arranged 4(M) x 2(N).
// Block tile: 128 batch rows x 32 hidden cols.  Wave tile: 32 x 16 per gate
// panel (2 A-frags reuse each B-frag -> 6 WMMA per K-step).
// Weight slices (96 gate-rows x K) are LDS-staged once per block and shared
// by all 4 M-waves (conflict-free padded strides).
// kx == 32  -> layer-0: x from fp32 [B,18] padded to K=32 (Xf).
// kx == 256 -> layer-1: x from bf16 [B,256] (Xb).
// ---------------------------------------------------------------------------
__global__ __launch_bounds__(256) void gru_step_kernel(
    const unsigned short* __restrict__ Xb,
    const float*          __restrict__ Xf,
    const unsigned short* __restrict__ Wih,   // bf16 [768, kx]
    const unsigned short* __restrict__ Whh,   // bf16 [768, 256]
    const float*          __restrict__ bih,   // f32 [768]
    const float*          __restrict__ bhh,   // f32 [768]
    const unsigned short* __restrict__ Hb,    // bf16 h_prev [B,256]
    const float*          __restrict__ Hf,    // f32  h_prev [B,256]
    unsigned short*       __restrict__ Ob,    // bf16 h_new
    float*                __restrict__ Of,    // f32  h_new
    int kx)
{
    extern __shared__ unsigned short smem[];
    unsigned short* sWhh = smem;                       // [96][LDSW_K256]
    unsigned short* sWih = smem + 96 * LDSW_K256;      // [96][LDSW_K256 or K32]

    const int tid  = threadIdx.x;
    const int lane = tid & 31;
    const int wave = tid >> 5;
    const int bn0  = blockIdx.y * 32;                  // block hidden-col base
    const int m0   = blockIdx.x * 128 + (wave >> 1) * 32;  // wave batch-row base
    const int wn16 = (wave & 1) * 16;                  // wave col offset in block

    // ---- stage weight slices into LDS (local row = gate*32 + colInBlock) --
    for (int c = tid; c < 96 * 16; c += 256) {         // 96 rows x 16 x 16B
        const int row = c >> 4, ch = c & 15;
        const int g = row >> 5, nl = row & 31;
        const size_t grow = (size_t)(g * GRU_H + bn0 + nl);
        *(u32x4*)(sWhh + row * LDSW_K256 + ch * 8) =
            *(const u32x4*)(Whh + grow * GRU_H + ch * 8);
    }
    if (kx == 32) {
        for (int c = tid; c < 96 * 2; c += 256) {      // 96 rows x 2 x 16B
            const int row = c >> 1, ch = c & 1;
            const int g = row >> 5, nl = row & 31;
            const size_t grow = (size_t)(g * GRU_H + bn0 + nl);
            *(u32x4*)(sWih + row * LDSW_K32 + ch * 8) =
                *(const u32x4*)(Wih + grow * 32 + ch * 8);
        }
    } else {
        for (int c = tid; c < 96 * 16; c += 256) {
            const int row = c >> 4, ch = c & 15;
            const int g = row >> 5, nl = row & 31;
            const size_t grow = (size_t)(g * GRU_H + bn0 + nl);
            *(u32x4*)(sWih + row * LDSW_K256 + ch * 8) =
                *(const u32x4*)(Wih + grow * GRU_H + ch * 8);
        }
    }
    __syncthreads();

    v8f accR[2]  = {};
    v8f accZ[2]  = {};
    v8f accXN[2] = {};
    v8f accHN[2] = {};

    // ---- hidden-state projection: h_prev @ Whh^T, K = 256 -----------------
#pragma unroll
    for (int k0 = 0; k0 < GRU_H; k0 += 32) {
        v16bf a0 = load_a_frag(Hb + (size_t)m0 * GRU_H + k0, GRU_H, lane);
        v16bf a1 = load_a_frag(Hb + (size_t)(m0 + 16) * GRU_H + k0, GRU_H, lane);
        v16bf br = load_b_frag_lds(sWhh, LDSW_K256,      wn16, k0, lane);
        v16bf bz = load_b_frag_lds(sWhh, LDSW_K256, 32 + wn16, k0, lane);
        v16bf bn = load_b_frag_lds(sWhh, LDSW_K256, 64 + wn16, k0, lane);
        accR[0]  = wmma_bf16(a0, br, accR[0]);
        accR[1]  = wmma_bf16(a1, br, accR[1]);
        accZ[0]  = wmma_bf16(a0, bz, accZ[0]);
        accZ[1]  = wmma_bf16(a1, bz, accZ[1]);
        accHN[0] = wmma_bf16(a0, bn, accHN[0]);
        accHN[1] = wmma_bf16(a1, bn, accHN[1]);
    }

    // ---- input projection: x @ Wih^T --------------------------------------
    if (kx == 32) {
        v16bf a0 = load_a_frag_x18(Xf, m0, lane);
        v16bf a1 = load_a_frag_x18(Xf, m0 + 16, lane);
        v16bf br = load_b_frag_lds(sWih, LDSW_K32,      wn16, 0, lane);
        v16bf bz = load_b_frag_lds(sWih, LDSW_K32, 32 + wn16, 0, lane);
        v16bf bn = load_b_frag_lds(sWih, LDSW_K32, 64 + wn16, 0, lane);
        accR[0]  = wmma_bf16(a0, br, accR[0]);
        accR[1]  = wmma_bf16(a1, br, accR[1]);
        accZ[0]  = wmma_bf16(a0, bz, accZ[0]);
        accZ[1]  = wmma_bf16(a1, bz, accZ[1]);
        accXN[0] = wmma_bf16(a0, bn, accXN[0]);
        accXN[1] = wmma_bf16(a1, bn, accXN[1]);
    } else {
#pragma unroll
        for (int k0 = 0; k0 < GRU_H; k0 += 32) {
            v16bf a0 = load_a_frag(Xb + (size_t)m0 * GRU_H + k0, GRU_H, lane);
            v16bf a1 = load_a_frag(Xb + (size_t)(m0 + 16) * GRU_H + k0, GRU_H, lane);
            v16bf br = load_b_frag_lds(sWih, LDSW_K256,      wn16, k0, lane);
            v16bf bz = load_b_frag_lds(sWih, LDSW_K256, 32 + wn16, k0, lane);
            v16bf bn = load_b_frag_lds(sWih, LDSW_K256, 64 + wn16, k0, lane);
            accR[0]  = wmma_bf16(a0, br, accR[0]);
            accR[1]  = wmma_bf16(a1, br, accR[1]);
            accZ[0]  = wmma_bf16(a0, bz, accZ[0]);
            accZ[1]  = wmma_bf16(a1, bz, accZ[1]);
            accXN[0] = wmma_bf16(a0, bn, accXN[0]);
            accXN[1] = wmma_bf16(a1, bn, accXN[1]);
        }
    }

    // ---- fused gate nonlinearities + state update, in C/D register layout -
    // v8f element e <-> (M = tileBase + e + 8*(lane>=16), N = col)
    const int   col  = bn0 + wn16 + (lane & 15);
    const float br_  = bih[col]             + bhh[col];
    const float bz_  = bih[GRU_H + col]     + bhh[GRU_H + col];
    const float bin_ = bih[2 * GRU_H + col];
    const float bhn_ = bhh[2 * GRU_H + col];

#pragma unroll
    for (int tIdx = 0; tIdx < 2; ++tIdx) {
        const int rbase = m0 + tIdx * 16 + (lane >> 4) * 8;
#pragma unroll
        for (int e = 0; e < 8; ++e) {
            const int   row = rbase + e;
            const float r   = sig_fast(accR[tIdx][e] + br_);
            const float z   = sig_fast(accZ[tIdx][e] + bz_);
            const float n   = tanh_fast(accXN[tIdx][e] + bin_ +
                                        r * (accHN[tIdx][e] + bhn_));
            const float hp  = Hf[(size_t)row * GRU_H + col];
            const float hv  = (1.0f - z) * n + z * hp;
            Of[(size_t)row * GRU_H + col] = hv;
            Ob[(size_t)row * GRU_H + col] = f2bf(hv);
        }
    }
}

// ---------------------------------------------------------------------------
// FC head: out = sigmoid(h2_last @ fc_w^T + fc_b) * 2 - 1 = tanh(logit/2)
// ---------------------------------------------------------------------------
__global__ __launch_bounds__(256) void gru_head_kernel(
    const float* __restrict__ h,      // f32 [B,256]
    const float* __restrict__ w,      // f32 [4,256]
    const float* __restrict__ bias,   // f32 [4]
    float*       __restrict__ out)    // f32 [B,4]
{
    const int b = blockIdx.x * blockDim.x + threadIdx.x;
    if (b >= GRU_B) return;
    float a0 = bias[0], a1 = bias[1], a2 = bias[2], a3 = bias[3];
    const float* hr = h + (size_t)b * GRU_H;
#pragma unroll 4
    for (int k = 0; k < GRU_H; ++k) {
        const float hv = hr[k];
        a0 += hv * w[k];
        a1 += hv * w[GRU_H + k];
        a2 += hv * w[2 * GRU_H + k];
        a3 += hv * w[3 * GRU_H + k];
    }
    out[b * 4 + 0] = tanh_fast(0.5f * a0);
    out[b * 4 + 1] = tanh_fast(0.5f * a1);
    out[b * 4 + 2] = tanh_fast(0.5f * a2);
    out[b * 4 + 3] = tanh_fast(0.5f * a3);
}

// ---------------------------------------------------------------------------
// Prep kernels
// ---------------------------------------------------------------------------
__global__ void cvt_bf16_kernel(const float* __restrict__ src,
                                unsigned short* __restrict__ dst, int n) {
    const int i = blockIdx.x * blockDim.x + threadIdx.x;
    if (i < n) dst[i] = f2bf(src[i]);
}

// fp32 [768,18] -> bf16 [768,32] zero-padded
__global__ void cvt_pad_kernel(const float* __restrict__ src,
                               unsigned short* __restrict__ dst) {
    const int i = blockIdx.x * blockDim.x + threadIdx.x;
    if (i < GRU_G * 32) {
        const int r = i >> 5, k = i & 31;
        dst[i] = (k < GRU_I) ? f2bf(src[r * GRU_I + k]) : (unsigned short)0;
    }
}

__global__ void zero_kernel(unsigned int* __restrict__ p, int n) {
    const int i = blockIdx.x * blockDim.x + threadIdx.x;
    if (i < n) p[i] = 0u;
}

// ---------------------------------------------------------------------------
extern "C" void kernel_launch(void* const* d_in, const int* in_sizes, int n_in,
                              void* d_out, int out_size, void* d_ws, size_t ws_size,
                              hipStream_t stream) {
    (void)in_sizes; (void)n_in; (void)out_size; (void)ws_size;

    const float* x     = (const float*)d_in[0];
    const float* W_ih0 = (const float*)d_in[1];
    const float* W_hh0 = (const float*)d_in[2];
    const float* b_ih0 = (const float*)d_in[3];
    const float* b_hh0 = (const float*)d_in[4];
    const float* W_ih1 = (const float*)d_in[5];
    const float* W_hh1 = (const float*)d_in[6];
    const float* b_ih1 = (const float*)d_in[7];
    const float* b_hh1 = (const float*)d_in[8];
    const float* fc_w  = (const float*)d_in[9];
    const float* fc_b  = (const float*)d_in[10];
    float* out = (float*)d_out;

    // ---- workspace layout (all offsets 256B-aligned) ----------------------
    char*  ws  = (char*)d_ws;
    size_t off = 0;
    auto take = [&](size_t bytes) -> char* {
        char* p = ws + off;
        off += (bytes + 255) & ~(size_t)255;
        return p;
    };
    unsigned short* wihb0 = (unsigned short*)take((size_t)GRU_G * 32    * 2);
    unsigned short* whhb0 = (unsigned short*)take((size_t)GRU_G * GRU_H * 2);
    unsigned short* wihb1 = (unsigned short*)take((size_t)GRU_G * GRU_H * 2);
    unsigned short* whhb1 = (unsigned short*)take((size_t)GRU_G * GRU_H * 2);
    const size_t hfB = (size_t)GRU_B * GRU_H * 4;
    const size_t hbB = (size_t)GRU_B * GRU_H * 2;
    float*          h1f[2] = { (float*)take(hfB),          (float*)take(hfB) };
    unsigned short* h1b[2] = { (unsigned short*)take(hbB), (unsigned short*)take(hbB) };
    float*          h2f[2] = { (float*)take(hfB),          (float*)take(hfB) };
    unsigned short* h2b[2] = { (unsigned short*)take(hbB), (unsigned short*)take(hbB) };

    // ---- prep: weight conversion + zero h0 --------------------------------
    cvt_pad_kernel<<<(GRU_G * 32 + 255) / 256, 256, 0, stream>>>(W_ih0, wihb0);
    cvt_bf16_kernel<<<(GRU_G * GRU_H + 255) / 256, 256, 0, stream>>>(W_hh0, whhb0, GRU_G * GRU_H);
    cvt_bf16_kernel<<<(GRU_G * GRU_H + 255) / 256, 256, 0, stream>>>(W_ih1, wihb1, GRU_G * GRU_H);
    cvt_bf16_kernel<<<(GRU_G * GRU_H + 255) / 256, 256, 0, stream>>>(W_hh1, whhb1, GRU_G * GRU_H);

    const int nf = GRU_B * GRU_H;          // f32 words
    const int nb = GRU_B * GRU_H / 2;      // bf16 buffer as u32 words
    zero_kernel<<<(nf + 255) / 256, 256, 0, stream>>>((unsigned int*)h1f[0], nf);
    zero_kernel<<<(nb + 255) / 256, 256, 0, stream>>>((unsigned int*)h1b[0], nb);
    zero_kernel<<<(nf + 255) / 256, 256, 0, stream>>>((unsigned int*)h2f[0], nf);
    zero_kernel<<<(nb + 255) / 256, 256, 0, stream>>>((unsigned int*)h2b[0], nb);

    // ---- recurrence: 2 fused layer-steps per timestep, ping-pong state ----
    const dim3   grid(GRU_B / 128, GRU_H / 32);               // (16, 8)
    const size_t smem0 = (size_t)(96 * LDSW_K256 + 96 * LDSW_K32)  * 2;  // 57 KB
    const size_t smem1 = (size_t)(96 * LDSW_K256 + 96 * LDSW_K256) * 2;  // 99 KB
    for (int t = 0; t < GRU_T; ++t) {
        const int cur = t & 1, nxt = cur ^ 1;
        const float* xt = x + (size_t)t * GRU_B * GRU_I;
        // layer 0: x_t (fp32, K padded to 32) + h1_prev -> h1_t
        gru_step_kernel<<<grid, 256, smem0, stream>>>(
            nullptr, xt, wihb0, whhb0, b_ih0, b_hh0,
            h1b[cur], h1f[cur], h1b[nxt], h1f[nxt], 32);
        // layer 1: h1_t (bf16, K=256) + h2_prev -> h2_t
        gru_step_kernel<<<grid, 256, smem1, stream>>>(
            h1b[nxt], nullptr, wihb1, whhb1, b_ih1, b_hh1,
            h2b[cur], h2f[cur], h2b[nxt], h2f[nxt], GRU_H);
    }

    // after t = 127 (cur=1, nxt=0) final h2 is in buffer 0
    gru_head_kernel<<<(GRU_B + 255) / 256, 256, 0, stream>>>(h2f[0], fc_w, fc_b, out);
}
